// InvariantPointAttention_88622355185762
// MI455X (gfx1250) — matrix operand order repaired and checked by reading, not compiled, not run on hardware
//
#include <hip/hip_runtime.h>
#include <hip/hip_bf16.h>

// ---------------------------------------------------------------------------
// Invariant Point Attention for MI455X (gfx1250, wave32, WMMA bf16 16x16x32)
// Memory-bound problem: inputs_2d = 512MB fp32 read twice (bias GEMM +
// res_2d GEMM) -> ~1.05GB HBM traffic -> ~45us floor at 23.3 TB/s.
// All contractions use v_wmma_f32_16x16x32_bf16; point-distance logits kept
// in fp32 VALU to avoid bf16 cancellation in ||q-k||^2.
// ---------------------------------------------------------------------------

typedef __bf16 bf16_t;
typedef __attribute__((ext_vector_type(16))) __bf16 v16bf;
typedef __attribute__((ext_vector_type(8)))  float  v8f;
typedef __attribute__((ext_vector_type(4)))  int    vi4;
typedef __attribute__((ext_vector_type(4)))  float  vf4;

#define NRES 1024
#define C1D  384
#define C2D  128
#define NH   12
#define FEAT 2112
#define NCH  384

// ---- workspace layout (bytes) ----
#define OFF_QS    0u                    // 1024*192 f32  = 786432
#define OFF_KS    786432u               // 1024*192 f32
#define OFF_QP    1572864u              // 1024*144 f32  = 589824
#define OFF_KP    2162688u              // 1024*144 f32
#define OFF_VSB   2752512u              // 12*16*1024 bf16 = 393216
#define OFF_VPB   3145728u              // 12*32*1024 bf16 = 786432
#define OFF_BIAS  3932160u              // 1024*1024*12 bf16 = 25165824
#define OFF_OS    29097984u             // 1024*192 f32
#define OFF_OP    29884416u             // 1024*288 f32 = 1179648
#define OFF_OR    31064064u             // 1024*1536 f32 = 6291456
#define OFF_W2DT  37355520u             // 16*128 bf16 = 4096
#define OFF_WOUTT 37359616u             // 384*2112 bf16 = 1622016
// end ~38.99 MB

__device__ __forceinline__ v8f wmma_bf(v16bf a, v16bf b, v8f c) {
  return __builtin_amdgcn_wmma_f32_16x16x32_bf16(false, a, false, b, (short)0, c,
                                                 false, false);
}

// A-fragment (16x32 bf16): lane L -> row (L&15); bytes (L&16?16:0)+{0,32}
__device__ __forceinline__ v16bf ldA(const char* base, int rowStrideB, int lane) {
  union { v16bf v; vi4 q[2]; } u;
  const char* r = base + (lane & 15) * rowStrideB + ((lane & 16) ? 16 : 0);
  u.q[0] = *(const vi4*)r;
  u.q[1] = *(const vi4*)(r + 32);
  return u.v;
}
// B-fragment (32x16 bf16): lane L -> col (L&15); bytes (L&16?16:0)+{0,32}
__device__ __forceinline__ v16bf ldB(const char* base, int colStrideB, int lane) {
  union { v16bf v; vi4 q[2]; } u;
  const char* r = base + (lane & 15) * colStrideB + ((lane & 16) ? 16 : 0);
  u.q[0] = *(const vi4*)r;
  u.q[1] = *(const vi4*)(r + 32);
  return u.v;
}

// ---------------------------------------------------------------------------
// Kernel 1: transpose/convert weights to bf16 B-operand layouts
// ---------------------------------------------------------------------------
__global__ void k_prep(const float* __restrict__ w2d, const float* __restrict__ wout,
                       bf16_t* __restrict__ w2dT, bf16_t* __restrict__ woutT) {
  long i = (long)blockIdx.x * 256 + threadIdx.x;
  if (i < 2048) {                       // w2dT[h(16 pad)][c(128)]
    int h = (int)(i / 128), c = (int)(i % 128);
    w2dT[i] = (h < NH) ? (bf16_t)w2d[c * NH + h] : (bf16_t)0.f;
  }
  long j = i - 2048;
  if (j >= 0 && j < (long)NCH * FEAT) { // woutT[o][f]
    long o = j / FEAT, f = j % FEAT;
    woutT[j] = (bf16_t)wout[f * NCH + o];
  }
}

// ---------------------------------------------------------------------------
// Kernel 2: projections + rigid transform (VALU; ~0.9 GFLOP total)
// ---------------------------------------------------------------------------
__global__ void k_proj(const float* __restrict__ x1d, const float* __restrict__ rot,
                       const float* __restrict__ trans,
                       const float* __restrict__ wqp, const float* __restrict__ bqp,
                       const float* __restrict__ wkp, const float* __restrict__ bkp,
                       const float* __restrict__ wvp, const float* __restrict__ bvp,
                       const float* __restrict__ wqs, const float* __restrict__ wks,
                       const float* __restrict__ wvs,
                       float* __restrict__ QS, float* __restrict__ KS,
                       float* __restrict__ QP, float* __restrict__ KP,
                       bf16_t* __restrict__ VSb, bf16_t* __restrict__ VPb) {
  __shared__ float x[C1D];
  int n = blockIdx.x, tid = threadIdx.x;
  for (int i = tid; i < C1D; i += 128) x[i] = x1d[(long)n * C1D + i];
  __syncthreads();
  const float* R = &rot[n * 9];
  const float* t = &trans[n * 3];
  for (int j = tid; j < 864; j += 128) {
    if (j < 576) {                                   // scalar q/k/v (192 each)
      int which = j / 192, o = j % 192;
      const float* w = (which == 0) ? wqs : (which == 1) ? wks : wvs;
      float acc = 0.f;
      for (int c = 0; c < C1D; ++c) acc += x[c] * w[c * 192 + o];
      if (which == 0)      QS[(long)n * 192 + o] = acc * 0.25f;   // * sqrt(1/16)
      else if (which == 1) KS[(long)n * 192 + o] = acc;
      else                 VSb[(long)o * NRES + n] = (bf16_t)acc; // [h*16+s][k]
    } else if (j < 672) {                            // q/k points (48 each)
      int jj = j - 576, which = jj / 48, hp = jj % 48, h = hp / 4, p = hp % 4;
      const float* w  = which ? wkp : wqp;
      const float* bb = which ? bkp : bqp;
      float l[3];
      for (int i = 0; i < 3; ++i) {
        int dd = i * 4 + p;                          // local[h][p][i]=proj[h][i*4+p]
        float acc = bb[h * 12 + dd];
        for (int c = 0; c < C1D; ++c) acc += x[c] * w[c * 144 + h * 12 + dd];
        l[i] = acc;
      }
      float* dst = which ? KP : QP;
      for (int i = 0; i < 3; ++i)
        dst[(long)n * 144 + h * 12 + p * 3 + i] =
            t[i] + R[i * 3 + 0] * l[0] + R[i * 3 + 1] * l[1] + R[i * 3 + 2] * l[2];
    } else if (j < 768) {                            // v points (96)
      int hp = j - 672, h = hp / 8, p = hp % 8;
      float l[3];
      for (int i = 0; i < 3; ++i) {
        int dd = i * 8 + p;
        float acc = bvp[h * 24 + dd];
        for (int c = 0; c < C1D; ++c) acc += x[c] * wvp[c * 288 + h * 24 + dd];
        l[i] = acc;
      }
      for (int i = 0; i < 3; ++i) {
        float g = t[i] + R[i * 3] * l[0] + R[i * 3 + 1] * l[1] + R[i * 3 + 2] * l[2];
        VPb[(long)(h * 32 + p * 3 + i) * NRES + n] = (bf16_t)g;  // [h][d(32 pad)][k]
      }
    } else {                                         // zero pad rows d=24..31
      int jj = j - 768, h = jj / 8, r = jj % 8;
      VPb[(long)(h * 32 + 24 + r) * NRES + n] = (bf16_t)0.f;
    }
  }
}

// ---------------------------------------------------------------------------
// Kernel 3: bias2d[qk,h] = inputs_2d[qk,:] @ w2d  (M=1M rows, K=128, N=16)
// One 16-row tile per wave; 4 chained bf16 WMMAs. Streams X once (512MB).
// ---------------------------------------------------------------------------
__global__ void k_bias(const float* __restrict__ X, const bf16_t* __restrict__ w2dT,
                       bf16_t* __restrict__ biasB) {
  int wv = threadIdx.x >> 5, lane = threadIdx.x & 31;
  long tile = (long)blockIdx.x * 8 + wv;
  long r0 = tile * 16;
  v16bf bfr[4];
#pragma unroll
  for (int t = 0; t < 4; ++t)
    bfr[t] = ldB((const char*)w2dT + t * 64, 256, lane);
  v8f acc = {};
#pragma unroll
  for (int t = 0; t < 4; ++t) {
    const float* rp = X + ((long)(r0 + (lane & 15))) * 128 + t * 32 + ((lane & 16) ? 8 : 0);
    vf4 f0 = *(const vf4*)rp;
    vf4 f1 = *(const vf4*)(rp + 4);
    vf4 f2 = *(const vf4*)(rp + 16);
    vf4 f3 = *(const vf4*)(rp + 20);
    v16bf a;
#pragma unroll
    for (int i = 0; i < 4; ++i) {
      a[i]      = (bf16_t)f0[i];
      a[4 + i]  = (bf16_t)f1[i];
      a[8 + i]  = (bf16_t)f2[i];
      a[12 + i] = (bf16_t)f3[i];
    }
    acc = wmma_bf(a, bfr[t], acc);
  }
  int hcol = lane & 15;
  if (hcol < NH) {
#pragma unroll
    for (int j = 0; j < 8; ++j) {
      long r = r0 + j + ((lane & 16) ? 8 : 0);
      biasB[r * NH + hcol] = (bf16_t)acc[j];
    }
  }
}

// ---------------------------------------------------------------------------
// Kernel 4: fused flash attention. 64 WGs (16 queries each), 8 wave32.
// Waves 0-3: 3 heads each -> fp32 logits + online softmax + P(bf16)->LDS,
//            then res_scalar/res_point WMMAs.
// Waves 4-7: stage X tile (bf16, [q][c][k] in LDS) + 4 queries' res_2d WMMAs.
// ---------------------------------------------------------------------------
__global__ void __launch_bounds__(256, 1) k_attn(
    const float* __restrict__ X2d, const float* __restrict__ maskg,
    const float* __restrict__ rawpw, const float* __restrict__ b2d,
    const float* __restrict__ QS, const float* __restrict__ KS,
    const float* __restrict__ QP, const float* __restrict__ KP,
    const bf16_t* __restrict__ VSb, const bf16_t* __restrict__ VPb,
    const bf16_t* __restrict__ biasB,
    float* __restrict__ outS, float* __restrict__ outP, float* __restrict__ outR) {
  extern __shared__ char smem[];
  bf16_t* Xb    = (bf16_t*)smem;                 // [16q][128c][32k] bf16 = 131072B
  bf16_t* Pb    = (bf16_t*)(smem + 131072);      // [16q][16h][32k]  bf16 = 16384B
  bf16_t* biasL = (bf16_t*)(smem + 147456);      // [16q][32k][12h]  bf16 = 12288B
  float*  qsL   = (float*)(smem + 159744);       // [16q][12h][16s]  = 12288B
  float*  qpL   = (float*)(smem + 172032);       // [16q][12h][12d]  = 9216B
  float*  scL   = (float*)(smem + 181248);       // [16q][12h] rescale
  float*  mL    = (float*)(smem + 182016);       // running max
  float*  lL    = (float*)(smem + 182784);       // running sum
  float*  mqL   = (float*)(smem + 183552);       // query mask

  int tid = threadIdx.x, wv = tid >> 5, lane = tid & 31;
  int half8 = (lane & 16) ? 8 : 0;
  int q0 = blockIdx.x * 16;

  for (int i = tid; i < 16 * 192; i += 256) {
    int q = i / 192; qsL[i] = QS[(long)(q0 + q) * 192 + (i % 192)];
  }
  for (int i = tid; i < 16 * 144; i += 256) {
    int q = i / 144; qpL[i] = QP[(long)(q0 + q) * 144 + (i % 144)];
  }
  if (tid < 16) mqL[tid] = maskg[q0 + tid];
  for (int i = tid; i < 192; i += 256) { mL[i] = -1e30f; lL[i] = 0.f; scL[i] = 0.f; }
  __syncthreads();

  float pw[3] = {0.f, 0.f, 0.f}, bh[3] = {0.f, 0.f, 0.f};
  if (wv < 4) {
#pragma unroll
    for (int hh = 0; hh < 3; ++hh) {
      int h = wv * 3 + hh;
      pw[hh] = 0.23570226f * log1pf(__expf(rawpw[h]));  // sqrt(1/18)*softplus
      bh[hh] = b2d[h];
    }
  }

  v8f accS[3]    = {};
  v8f accP[3][2] = {};
  v8f acc2[4][8] = {};

  for (int kt = 0; kt < 32; ++kt) {
    int k0 = kt * 32;
    // ---------------- phase A: staging ----------------
    if (wv >= 4) {
      int t2 = tid - 128;                 // 0..127
      int q = t2 >> 3, k4 = (t2 & 7) * 4;
      const float* rp = X2d + ((long)(q0 + q) * NRES + k0 + k4) * C2D;
      bf16_t* dst = Xb + (q * 128) * 32 + k4;
      for (int c = 0; c < 128; c += 4) {
        vf4 a0 = *(const vf4*)(rp + 0 * 128 + c);
        vf4 a1 = *(const vf4*)(rp + 1 * 128 + c);
        vf4 a2 = *(const vf4*)(rp + 2 * 128 + c);
        vf4 a3 = *(const vf4*)(rp + 3 * 128 + c);
#pragma unroll
        for (int cc = 0; cc < 4; ++cc) {
          union { bf16_t h[4]; unsigned long long u; } pk;
          pk.h[0] = (bf16_t)a0[cc]; pk.h[1] = (bf16_t)a1[cc];
          pk.h[2] = (bf16_t)a2[cc]; pk.h[3] = (bf16_t)a3[cc];
          *(unsigned long long*)(dst + (c + cc) * 32) = pk.u;
        }
      }
      if (kt < 31) {                       // prefetch next k-tile (gfx1250)
        __builtin_prefetch(rp + 32 * C2D, 0, 1);
        __builtin_prefetch(rp + 34 * C2D, 0, 1);
      }
    } else {
      // bias tile (16q x 32k x 12h bf16 = 12KB) -> LDS
#pragma unroll
      for (int i = 0; i < 6; ++i) {
        int idx = (tid * 6 + i) * 16;
        int qq = idx / 768, rem = idx % 768;
        vi4 v = *(const vi4*)((const char*)biasB +
                              ((long)(q0 + qq) * NRES + k0) * 24 + rem);
        *(vi4*)((char*)biasL + idx) = v;
      }
    }
    __syncthreads();
    // ---------------- phase B: logits + online softmax ----------------
    if (wv < 4) {
#pragma unroll
      for (int hh = 0; hh < 3; ++hh) {
        int h = wv * 3 + hh;
        float c01[2][8];
#pragma unroll
        for (int tI = 0; tI < 2; ++tI) {
          int k = k0 + tI * 16 + (lane & 15);
          const vf4* ksp = (const vf4*)(KS + (long)k * 192 + h * 16);
          vf4 ksv[4]; ksv[0]=ksp[0]; ksv[1]=ksp[1]; ksv[2]=ksp[2]; ksv[3]=ksp[3];
          const vf4* kpp = (const vf4*)(KP + (long)k * 144 + h * 12);
          vf4 kpv[3]; kpv[0]=kpp[0]; kpv[1]=kpp[1]; kpv[2]=kpp[2];
          float mk = maskg[k];
#pragma unroll
          for (int j = 0; j < 8; ++j) {
            int q = j + half8;
            const vf4* qsp = (const vf4*)(qsL + (q * 12 + h) * 16);
            const vf4* qpp = (const vf4*)(qpL + (q * 12 + h) * 12);
            float dot = 0.f, d2 = 0.f;
#pragma unroll
            for (int ii = 0; ii < 4; ++ii) {
              vf4 a = qsp[ii], b = ksv[ii];
              dot += a[0]*b[0] + a[1]*b[1] + a[2]*b[2] + a[3]*b[3];
            }
#pragma unroll
            for (int ii = 0; ii < 3; ++ii) {
              vf4 d = qpp[ii] - kpv[ii];
              d2 += d[0]*d[0] + d[1]*d[1] + d[2]*d[2] + d[3]*d[3];
            }
            float lg = dot - 0.5f * pw[hh] * d2 + bh[hh] +
                       (float)biasL[(q * 32 + tI * 16 + (lane & 15)) * 12 + h];
            lg = (lg - 1e5f * (1.f - mqL[q] * mk)) * 0.57735027f;
            c01[tI][j] = lg;
          }
        }
#pragma unroll
        for (int j = 0; j < 8; ++j) {
          int q = j + half8;
          float vmax = fmaxf(c01[0][j], c01[1][j]);
          vmax = fmaxf(vmax, __shfl_xor(vmax, 1));
          vmax = fmaxf(vmax, __shfl_xor(vmax, 2));
          vmax = fmaxf(vmax, __shfl_xor(vmax, 4));
          vmax = fmaxf(vmax, __shfl_xor(vmax, 8));
          float mo = mL[q * 12 + h], lo = lL[q * 12 + h];
          float mn = fmaxf(mo, vmax);
          float sc = __expf(mo - mn);
          float p0 = __expf(c01[0][j] - mn);
          float p1 = __expf(c01[1][j] - mn);
          float rs = p0 + p1;
          rs += __shfl_xor(rs, 1);
          rs += __shfl_xor(rs, 2);
          rs += __shfl_xor(rs, 4);
          rs += __shfl_xor(rs, 8);
          if ((lane & 15) == 0) {
            mL[q * 12 + h] = mn;
            lL[q * 12 + h] = lo * sc + rs;
            scL[q * 12 + h] = sc;
          }
          Pb[(q * 16 + h) * 32 + (lane & 15)]      = (bf16_t)p0;
          Pb[(q * 16 + h) * 32 + 16 + (lane & 15)] = (bf16_t)p1;
        }
      }
    }
    __syncthreads();
    // ---------------- phase C: accumulation WMMAs ----------------
    if (wv < 4) {
#pragma unroll
      for (int hh = 0; hh < 3; ++hh) {
        int h = wv * 3 + hh;
#pragma unroll
        for (int j = 0; j < 8; ++j) {
          float f = scL[(j + half8) * 12 + h];
          accS[hh][j]    *= f;
          accP[hh][0][j] *= f;
          accP[hh][1][j] *= f;
        }
        v16bf a = ldA((const char*)Pb + h * 64, 1024, lane);   // rows = q
        v16bf b = ldB((const char*)VSb + ((long)h * 16 * NRES + k0) * 2, 2048, lane);
        accS[hh] = wmma_bf(a, b, accS[hh]);
#pragma unroll
        for (int ct = 0; ct < 2; ++ct) {
          v16bf bp = ldB((const char*)VPb + ((long)(h * 32 + ct * 16) * NRES + k0) * 2,
                         2048, lane);
          accP[hh][ct] = wmma_bf(a, bp, accP[hh][ct]);
        }
      }
    } else {
      int qb = (wv - 4) * 4;
#pragma unroll
      for (int u = 0; u < 4; ++u) {
        int qi = qb + u;
#pragma unroll
        for (int j = 0; j < 8; ++j) {
          int hr = j + half8;
          float f = (hr < NH) ? scL[qi * 12 + hr] : 1.0f;
#pragma unroll
          for (int ct = 0; ct < 8; ++ct) acc2[u][ct][j] *= f;
        }
        v16bf a = ldA((const char*)Pb + qi * 1024, 64, lane);  // rows = h
#pragma unroll
        for (int ct = 0; ct < 8; ++ct) {
          v16bf b = ldB((const char*)Xb + (qi * 128 + ct * 16) * 64, 64, lane);
          acc2[u][ct] = wmma_bf(a, b, acc2[u][ct]);
        }
      }
    }
    __syncthreads();
  }
  // ---------------- finalize: divide by softmax denominator ----------------
  if (wv < 4) {
#pragma unroll
    for (int hh = 0; hh < 3; ++hh) {
      int h = wv * 3 + hh;
#pragma unroll
      for (int j = 0; j < 8; ++j) {
        int q = j + half8;
        float inv = 1.0f / lL[q * 12 + h];
        int d0 = lane & 15;
        outS[((long)(q0 + q) * NH + h) * 16 + d0] = accS[hh][j] * inv;
        outP[((long)(q0 + q) * NH + h) * 24 + d0] = accP[hh][0][j] * inv;
        if (d0 < 8)
          outP[((long)(q0 + q) * NH + h) * 24 + 16 + d0] = accP[hh][1][j] * inv;
      }
    }
  } else {
    int qb = (wv - 4) * 4;
#pragma unroll
    for (int u = 0; u < 4; ++u) {
      int qi = qb + u;
#pragma unroll
      for (int j = 0; j < 8; ++j) {
        int hr = j + half8;
        if (hr < NH) {
          float inv = 1.0f / lL[qi * 12 + hr];
#pragma unroll
          for (int ct = 0; ct < 8; ++ct)
            outR[((long)(q0 + qi) * NH + hr) * C2D + ct * 16 + (lane & 15)] =
                acc2[u][ct][j] * inv;
        }
      }
    }
  }
}

// ---------------------------------------------------------------------------
// Kernel 5: feature assembly (inverse rigid + norms) + output GEMM
// (16 x 2112) @ (2112 x 384): 66-deep bf16 WMMA chain, 24 column tiles.
// ---------------------------------------------------------------------------
__global__ void k_out(const float* __restrict__ outS, const float* __restrict__ outP,
                      const float* __restrict__ outR, const float* __restrict__ rot,
                      const float* __restrict__ trans, const float* __restrict__ bout,
                      const bf16_t* __restrict__ woutT, float* __restrict__ out) {
  extern __shared__ char smem[];
  bf16_t* fb = (bf16_t*)smem;            // [16][2112] bf16 = 67584B
  int tid = threadIdx.x, q0 = blockIdx.x * 16;
  for (int i = tid; i < 16 * 192; i += 256) {
    int q = i / 192, r = i % 192;
    fb[q * FEAT + r] = (bf16_t)outS[(long)(q0 + q) * 192 + r];
  }
  for (int i = tid; i < 16 * 1536; i += 256) {
    int q = i / 1536, r = i % 1536;
    fb[q * FEAT + 576 + r] = (bf16_t)outR[(long)(q0 + q) * 1536 + r];
  }
  for (int i = tid; i < 16 * 96; i += 256) {
    int q = i / 96, hp = i % 96, h = hp / 8, p = hp % 8;
    const float* g = &outP[((long)(q0 + q) * NH + h) * 24 + p * 3];
    const float* R = &rot[(q0 + q) * 9];
    const float* t = &trans[(q0 + q) * 3];
    float gx = g[0] - t[0], gy = g[1] - t[1], gz = g[2] - t[2];
    float lx = R[0] * gx + R[3] * gy + R[6] * gz;   // R^T (g - t)
    float ly = R[1] * gx + R[4] * gy + R[7] * gz;
    float lz = R[2] * gx + R[5] * gy + R[8] * gz;
    float nm = sqrtf(fmaxf(lx * lx + ly * ly + lz * lz, 1e-16f));
    fb[q * FEAT + 192 + hp] = (bf16_t)lx;
    fb[q * FEAT + 288 + hp] = (bf16_t)ly;
    fb[q * FEAT + 384 + hp] = (bf16_t)lz;
    fb[q * FEAT + 480 + hp] = (bf16_t)nm;
  }
  __syncthreads();
  int wv = tid >> 5, lane = tid & 31, half8 = (lane & 16) ? 8 : 0;
  v8f acc[3] = {};
  for (int k0 = 0; k0 < FEAT; k0 += 32) {
    v16bf a = ldA((const char*)fb + k0 * 2, FEAT * 2, lane);
#pragma unroll
    for (int nt = 0; nt < 3; ++nt) {
      long col0 = (wv * 3 + nt) * 16;
      v16bf b = ldB((const char*)woutT + col0 * (FEAT * 2) + k0 * 2, FEAT * 2, lane);
      acc[nt] = wmma_bf(a, b, acc[nt]);
    }
  }
#pragma unroll
  for (int nt = 0; nt < 3; ++nt) {
    int col = (wv * 3 + nt) * 16 + (lane & 15);
    float bo = bout[col];
#pragma unroll
    for (int j = 0; j < 8; ++j)
      out[(long)(q0 + j + half8) * NCH + col] = acc[nt][j] + bo;
  }
}

// ---------------------------------------------------------------------------
extern "C" void kernel_launch(void* const* d_in, const int* in_sizes, int n_in,
                              void* d_out, int out_size, void* d_ws, size_t ws_size,
                              hipStream_t stream) {
  const float* in1d  = (const float*)d_in[0];
  const float* in2d  = (const float*)d_in[1];
  const float* maskg = (const float*)d_in[2];
  const float* rot   = (const float*)d_in[3];
  const float* trans = (const float*)d_in[4];
  const float* rawpw = (const float*)d_in[5];
  const float* wqp   = (const float*)d_in[6];
  const float* bqp   = (const float*)d_in[7];
  const float* wkp   = (const float*)d_in[8];
  const float* bkp   = (const float*)d_in[9];
  const float* wvp   = (const float*)d_in[10];
  const float* bvp   = (const float*)d_in[11];
  const float* wqs   = (const float*)d_in[12];
  const float* wks   = (const float*)d_in[13];
  const float* wvs   = (const float*)d_in[14];
  const float* w2d   = (const float*)d_in[15];
  const float* b2d   = (const float*)d_in[16];
  const float* wout  = (const float*)d_in[17];
  const float* bout  = (const float*)d_in[18];

  char* ws = (char*)d_ws;
  float*  QS    = (float*)(ws + OFF_QS);
  float*  KS    = (float*)(ws + OFF_KS);
  float*  QP    = (float*)(ws + OFF_QP);
  float*  KP    = (float*)(ws + OFF_KP);
  bf16_t* VSb   = (bf16_t*)(ws + OFF_VSB);
  bf16_t* VPb   = (bf16_t*)(ws + OFF_VPB);
  bf16_t* biasB = (bf16_t*)(ws + OFF_BIAS);
  float*  oS    = (float*)(ws + OFF_OS);
  float*  oP    = (float*)(ws + OFF_OP);
  float*  oR    = (float*)(ws + OFF_OR);
  bf16_t* w2dT  = (bf16_t*)(ws + OFF_W2DT);
  bf16_t* woutT = (bf16_t*)(ws + OFF_WOUTT);

  k_prep<<<(2048 + NCH * FEAT + 255) / 256, 256, 0, stream>>>(w2d, wout, w2dT, woutT);
  k_proj<<<NRES, 128, 0, stream>>>(in1d, rot, trans, wqp, bqp, wkp, bkp, wvp, bvp,
                                   wqs, wks, wvs, QS, KS, QP, KP, VSb, VPb);
  k_bias<<<(NRES * NRES / 16) / 8, 256, 0, stream>>>(in2d, w2dT, biasB);
  k_attn<<<NRES / 16, 256, 183616, stream>>>(in2d, maskg, rawpw, b2d, QS, KS, QP, KP,
                                             VSb, VPb, biasB, oS, oP, oR);
  k_out<<<NRES / 16, 256, 16 * FEAT * 2, stream>>>(oS, oP, oR, rot, trans, bout,
                                                   woutT, (float*)d_out);
}
